// MoEFFN_5832565588003
// MI455X (gfx1250) — compile-verified, matrix operand order checked by reading, not run on hardware
//
#include <hip/hip_runtime.h>
#include <hip/hip_bf16.h>
#include <stdint.h>

// ---------------- problem constants ----------------
#define DIMX  1024      // model dim
#define EDIM  512       // expert hidden dim
#define NEXP  16
#define NTOK  4096      // B*T
#define CAPX  4096      // per-expert slot capacity (worst case)
#define NSLOT (NTOK*2)  // token*2+k slots
#define SHH   1024      // shared expert hidden

// ---------------- GEMM tile constants ----------------
#define MT  128         // M tile (rows / tokens)
#define NT  64          // N tile (output cols per block)
#define KT  32          // K per WMMA step (bf16)
#define LDA 40          // LDS row stride in u16 (32 + 8 pad, keeps 16B align)

typedef __attribute__((ext_vector_type(16))) __bf16 bf16x16;
typedef __attribute__((ext_vector_type(8)))  float  f32x8;
typedef __attribute__((ext_vector_type(4)))  unsigned int u32x4;
typedef int v4i __attribute__((vector_size(4 * sizeof(int))));

#define AS1 __attribute__((address_space(1)))
#define AS3 __attribute__((address_space(3)))

union FragU { u32x4 u[2]; bf16x16 v; };
union H8   { unsigned short s[8]; u32x4 v; };

// pack two fp32 -> {bf16(a) lo, bf16(b) hi} with one v_perm_b32 (+RNE-ish bias)
__device__ __forceinline__ unsigned pkbf(float a, float b) {
  unsigned ua = __float_as_uint(a) + 0x7fffu;
  unsigned ub = __float_as_uint(b) + 0x7fffu;
  // src0 = ub (combined bytes 4..7), src1 = ua (combined bytes 0..3)
  return __builtin_amdgcn_perm(ub, ua, 0x07060302u);
}

__device__ __forceinline__ unsigned short f2bf(float f) {
  unsigned u = __float_as_uint(f);
  u += 0x7fffu + ((u >> 16) & 1u);
  return (unsigned short)(u >> 16);
}

// ---- CDNA5 async global->LDS copy (ASYNCcnt) with sync fallback ----
__device__ __forceinline__ void cp16_async(void* lds, const void* g) {
#if __has_builtin(__builtin_amdgcn_global_load_async_to_lds_b128)
  __builtin_amdgcn_global_load_async_to_lds_b128(
      (AS1 v4i*)(uintptr_t)g,
      (AS3 v4i*)(unsigned)(uintptr_t)lds,
      0, 0);
#else
  *(u32x4*)lds = *(const u32x4*)g;
#endif
}

__device__ __forceinline__ void wait_async() {
#if __has_builtin(__builtin_amdgcn_s_wait_asynccnt)
  __builtin_amdgcn_s_wait_asynccnt(0);
#elif __has_builtin(__builtin_amdgcn_global_load_async_to_lds_b128)
  asm volatile("s_wait_asynccnt 0" ::: "memory");
#endif
}

// Load one 16x32 bf16 WMMA operand fragment from LDS.
// Layout (ISA 7.12.2): lane L holds row/col L%16;
// elems 0..7 -> K = kh..kh+7, elems 8..15 -> K = 16+kh..16+kh+7, kh = 8*(L/16).
__device__ __forceinline__ bf16x16 ldsFrag(const unsigned short* base, int row0, int lane) {
  const unsigned short* p = base + (row0 + (lane & 15)) * LDA + ((lane >> 4) << 3);
  FragU f;
  f.u[0] = *(const u32x4*)(p);
  f.u[1] = *(const u32x4*)(p + 16);
  return f.v;
}

__device__ __forceinline__ f32x8 wmma_bf16(bf16x16 a, bf16x16 b, f32x8 c) {
  return __builtin_amdgcn_wmma_f32_16x16x32_bf16(false, a, false, b, (short)0, c,
                                                 false, false);
}

__device__ __forceinline__ float silu(float g) {
  return g / (1.f + __expf(-g));
}

// =====================================================================
// Kernel 0: zero counters + zero-page
// =====================================================================
__global__ void k_init(int* __restrict__ cnt, unsigned* __restrict__ zbuf) {
  if (threadIdx.x < NEXP) cnt[threadIdx.x] = 0;
  if (threadIdx.x < 32)   zbuf[threadIdx.x] = 0u;   // 128 B zero page
}

// =====================================================================
// Kernel 1: router — logits, top-2, softmax, bucket tokens per expert
// =====================================================================
__global__ __launch_bounds__(256)
void k_router(const float* __restrict__ x,
              const float* __restrict__ rw,
              const float* __restrict__ rb,
              int*   __restrict__ cnt,
              int*   __restrict__ slotlist,   // [NEXP][CAPX]
              float* __restrict__ wslot) {    // [NSLOT]
  int n = blockIdx.x * blockDim.x + threadIdx.x;
  if (n >= NTOK) return;
  float lg[NEXP];
#pragma unroll
  for (int e = 0; e < NEXP; ++e) lg[e] = rb[e];
  const float* xr = x + (size_t)n * DIMX;
  for (int k = 0; k < DIMX; k += 4) {
    float4 xv = *(const float4*)(xr + k);
#pragma unroll
    for (int e = 0; e < NEXP; ++e) {
      const float* wr = rw + e * DIMX + k;
      lg[e] += xv.x * wr[0] + xv.y * wr[1] + xv.z * wr[2] + xv.w * wr[3];
    }
  }
  int i0 = 0; float b0 = lg[0];
#pragma unroll
  for (int e = 1; e < NEXP; ++e) if (lg[e] > b0) { b0 = lg[e]; i0 = e; }
  int i1 = (i0 == 0) ? 1 : 0; float b1 = lg[i1];
#pragma unroll
  for (int e = 0; e < NEXP; ++e)
    if (e != i0 && lg[e] > b1) { b1 = lg[e]; i1 = e; }
  float m  = fmaxf(b0, b1);
  float e0 = __expf(b0 - m), e1 = __expf(b1 - m);
  float inv = 1.f / (e0 + e1);
  int   ids[2] = { i0, i1 };
  float ws [2] = { e0 * inv, e1 * inv };
#pragma unroll
  for (int k = 0; k < 2; ++k) {
    int e    = ids[k];
    int pos  = atomicAdd(&cnt[e], 1);
    int slot = n * 2 + k;
    slotlist[e * CAPX + pos] = slot;
    wslot[slot] = ws[k];
  }
}

// =====================================================================
// Kernel 2a: flat fp32 -> bf16 (layout preserved), 8 elems/thread
// =====================================================================
__global__ __launch_bounds__(256)
void k_cvt(const float* __restrict__ src, unsigned short* __restrict__ dst,
           long n) {
  long i = ((long)blockIdx.x * 256 + threadIdx.x) * 8;
  if (i >= n) return;
  float4 a = *(const float4*)(src + i);
  float4 b = *(const float4*)(src + i + 4);
  u32x4 o = { pkbf(a.x, a.y), pkbf(a.z, a.w), pkbf(b.x, b.y), pkbf(b.z, b.w) };
  *(u32x4*)(dst + i) = o;
}

// =====================================================================
// Kernel 2b: transpose-convert fp32 [K][N] -> bf16 [N][K] (per expert)
// grid: (N/64, K/64, E), block 256
// =====================================================================
__global__ __launch_bounds__(256)
void k_cvt_t(const float* __restrict__ src, unsigned short* __restrict__ dst,
             int K, int N) {
  int e = blockIdx.z;
  src += (size_t)e * K * N;
  dst += (size_t)e * K * N;
  int n0 = blockIdx.x * 64, k0 = blockIdx.y * 64;
  __shared__ unsigned short Ts[64 * 68];   // [k][n], stride 68
  int t = threadIdx.x;
#pragma unroll
  for (int p = 0; p < 4; ++p) {
    int idx = t + p * 256;
    int k = idx >> 4, ns = (idx & 15) * 4;
    float4 v = *(const float4*)(src + (size_t)(k0 + k) * N + n0 + ns);
    *(uint2*)(&Ts[k * 68 + ns]) = make_uint2(pkbf(v.x, v.y), pkbf(v.z, v.w));
  }
  __syncthreads();
#pragma unroll
  for (int p = 0; p < 2; ++p) {
    int idx = t + p * 256;               // 512 chunks: 64 n-rows x 8 chunks
    int n = idx >> 3, kc = (idx & 7) * 8;
    H8 h;
#pragma unroll
    for (int q = 0; q < 8; ++q) h.s[q] = Ts[(kc + q) * 68 + n];
    *(u32x4*)(dst + (size_t)(n0 + n) * K + k0 + kc) = h.v;
  }
}

// =====================================================================
// Kernel 3: expert up+gate GEMM (gathered bf16 rows) + SwiGLU -> hidden
// grid: (EDIM/NT, CAPX/MT, NEXP), block 256 (8 waves, 4x2)
// =====================================================================
__global__ __launch_bounds__(256)
void k_upgate(const unsigned short* __restrict__ xb,    // bf16 [NTOK][DIMX]
              const unsigned short* __restrict__ gwt,   // bf16 [E][EDIM][DIMX]
              const unsigned short* __restrict__ uwt,   // bf16 [E][EDIM][DIMX]
              const int*   __restrict__ cnt,
              const int*   __restrict__ slotlist,
              const unsigned short* __restrict__ zbuf,
              unsigned short* __restrict__ hidden) {    // bf16 [NSLOT][EDIM]
  int e     = blockIdx.z;
  int count = cnt[e];
  int m0    = blockIdx.y * MT;
  if (m0 >= count) return;
  int ntile = blockIdx.x * NT;

  __shared__ unsigned short As[2][MT * LDA];
  __shared__ unsigned short Bg[2][NT * LDA];
  __shared__ unsigned short Bu[2][NT * LDA];
  __shared__ int slotS[MT];

  int t = threadIdx.x, lane = t & 31, w = t >> 5;
  int mW = (w >> 1) * 32, nW = (w & 1) * 32;

  if (t < MT) {
    int gm = m0 + t;
    slotS[t] = (gm < count) ? slotlist[e * CAPX + gm] : -1;
  }
  __syncthreads();

  const unsigned short* gwe = gwt + (size_t)e * EDIM * DIMX;
  const unsigned short* uwe = uwt + (size_t)e * EDIM * DIMX;

  auto stage = [&](int buf, int k0) {
#pragma unroll
    for (int p = 0; p < 2; ++p) {
      int idx = t + p * 256;
      int row = idx >> 2, c = idx & 3;
      int slot = slotS[row];
      const unsigned short* src = (slot >= 0)
          ? xb + (size_t)(slot >> 1) * DIMX + k0 + c * 8
          : zbuf + c * 8;
      cp16_async(&As[buf][row * LDA + c * 8], src);
    }
    {
      int row = t >> 2, c = t & 3;
      cp16_async(&Bg[buf][row * LDA + c * 8],
                 gwe + (size_t)(ntile + row) * DIMX + k0 + c * 8);
      cp16_async(&Bu[buf][row * LDA + c * 8],
                 uwe + (size_t)(ntile + row) * DIMX + k0 + c * 8);
    }
  };

  f32x8 zero = {0.f,0.f,0.f,0.f,0.f,0.f,0.f,0.f};
  f32x8 accG[2][2], accU[2][2];
#pragma unroll
  for (int i = 0; i < 2; ++i)
#pragma unroll
    for (int j = 0; j < 2; ++j) { accG[i][j] = zero; accU[i][j] = zero; }

  stage(0, 0);
  const int NIT = DIMX / KT;
  for (int it = 0; it < NIT; ++it) {
    wait_async();
    __syncthreads();
    if (it + 1 < NIT) stage((it + 1) & 1, (it + 1) * KT);
    int b = it & 1;
    bf16x16 a0 = ldsFrag(As[b], mW +  0, lane);
    bf16x16 a1 = ldsFrag(As[b], mW + 16, lane);
    bf16x16 g0 = ldsFrag(Bg[b], nW +  0, lane);
    bf16x16 g1 = ldsFrag(Bg[b], nW + 16, lane);
    bf16x16 u0 = ldsFrag(Bu[b], nW +  0, lane);
    bf16x16 u1 = ldsFrag(Bu[b], nW + 16, lane);
    accG[0][0] = wmma_bf16(a0, g0, accG[0][0]);
    accG[0][1] = wmma_bf16(a0, g1, accG[0][1]);
    accG[1][0] = wmma_bf16(a1, g0, accG[1][0]);
    accG[1][1] = wmma_bf16(a1, g1, accG[1][1]);
    accU[0][0] = wmma_bf16(a0, u0, accU[0][0]);
    accU[0][1] = wmma_bf16(a0, u1, accU[0][1]);
    accU[1][0] = wmma_bf16(a1, u0, accU[1][0]);
    accU[1][1] = wmma_bf16(a1, u1, accU[1][1]);
  }

  int rbase = 8 * (lane >> 4);
  int nlo   = lane & 15;
#pragma unroll
  for (int i = 0; i < 2; ++i)
#pragma unroll
    for (int j = 0; j < 2; ++j) {
      int col = ntile + nW + j * 16 + nlo;
#pragma unroll
      for (int v = 0; v < 8; ++v) {
        int m = mW + i * 16 + rbase + v;
        int slot = slotS[m];
        if (slot < 0) continue;
        float g = accG[i][j][v], u = accU[i][j][v];
        hidden[(size_t)slot * EDIM + col] = f2bf(silu(g) * u);
      }
    }
}

// =====================================================================
// Kernel 4: shared expert hidden = silu(x@shg^T)*(x@shu^T) -> bf16
// grid: (SHH/NT, NTOK/MT), block 256
// =====================================================================
__global__ __launch_bounds__(256)
void k_shhidden(const unsigned short* __restrict__ xb,    // bf16 [NTOK][DIMX]
                const unsigned short* __restrict__ shgb,  // bf16 [SHH][DIMX]
                const unsigned short* __restrict__ shub,  // bf16 [SHH][DIMX]
                unsigned short* __restrict__ shh) {       // bf16 [NTOK][SHH]
  int m0    = blockIdx.y * MT;
  int ntile = blockIdx.x * NT;

  __shared__ unsigned short As[2][MT * LDA];
  __shared__ unsigned short Bg[2][NT * LDA];
  __shared__ unsigned short Bu[2][NT * LDA];

  int t = threadIdx.x, lane = t & 31, w = t >> 5;
  int mW = (w >> 1) * 32, nW = (w & 1) * 32;

  auto stage = [&](int buf, int k0) {
#pragma unroll
    for (int p = 0; p < 2; ++p) {
      int idx = t + p * 256;
      int row = idx >> 2, c = idx & 3;
      cp16_async(&As[buf][row * LDA + c * 8],
                 xb + (size_t)(m0 + row) * DIMX + k0 + c * 8);
    }
    {
      int row = t >> 2, c = t & 3;
      cp16_async(&Bg[buf][row * LDA + c * 8],
                 shgb + (size_t)(ntile + row) * DIMX + k0 + c * 8);
      cp16_async(&Bu[buf][row * LDA + c * 8],
                 shub + (size_t)(ntile + row) * DIMX + k0 + c * 8);
    }
  };

  f32x8 zero = {0.f,0.f,0.f,0.f,0.f,0.f,0.f,0.f};
  f32x8 accG[2][2], accU[2][2];
#pragma unroll
  for (int i = 0; i < 2; ++i)
#pragma unroll
    for (int j = 0; j < 2; ++j) { accG[i][j] = zero; accU[i][j] = zero; }

  stage(0, 0);
  const int NIT = DIMX / KT;
  for (int it = 0; it < NIT; ++it) {
    wait_async();
    __syncthreads();
    if (it + 1 < NIT) stage((it + 1) & 1, (it + 1) * KT);
    int b = it & 1;
    bf16x16 a0 = ldsFrag(As[b], mW +  0, lane);
    bf16x16 a1 = ldsFrag(As[b], mW + 16, lane);
    bf16x16 g0 = ldsFrag(Bg[b], nW +  0, lane);
    bf16x16 g1 = ldsFrag(Bg[b], nW + 16, lane);
    bf16x16 u0 = ldsFrag(Bu[b], nW +  0, lane);
    bf16x16 u1 = ldsFrag(Bu[b], nW + 16, lane);
    accG[0][0] = wmma_bf16(a0, g0, accG[0][0]);
    accG[0][1] = wmma_bf16(a0, g1, accG[0][1]);
    accG[1][0] = wmma_bf16(a1, g0, accG[1][0]);
    accG[1][1] = wmma_bf16(a1, g1, accG[1][1]);
    accU[0][0] = wmma_bf16(a0, u0, accU[0][0]);
    accU[0][1] = wmma_bf16(a0, u1, accU[0][1]);
    accU[1][0] = wmma_bf16(a1, u0, accU[1][0]);
    accU[1][1] = wmma_bf16(a1, u1, accU[1][1]);
  }

  int rbase = 8 * (lane >> 4);
  int nlo   = lane & 15;
#pragma unroll
  for (int i = 0; i < 2; ++i)
#pragma unroll
    for (int j = 0; j < 2; ++j) {
      int col = ntile + nW + j * 16 + nlo;
#pragma unroll
      for (int v = 0; v < 8; ++v) {
        int m = m0 + mW + i * 16 + rbase + v;
        float g = accG[i][j][v], u = accU[i][j][v];
        shh[(size_t)m * SHH + col] = f2bf(silu(g) * u);
      }
    }
}

// =====================================================================
// Kernel 5: shared down: out = shh @ shd^T  (writes out, no atomics)
// grid: (DIMX/NT, NTOK/MT), block 256
// =====================================================================
__global__ __launch_bounds__(256)
void k_shdown(const unsigned short* __restrict__ shh,   // bf16 [NTOK][SHH]
              const unsigned short* __restrict__ shdb,  // bf16 [DIMX][SHH]
              float* __restrict__ out) {
  int m0    = blockIdx.y * MT;
  int ntile = blockIdx.x * NT;

  __shared__ unsigned short As[2][MT * LDA];
  __shared__ unsigned short Bs[2][NT * LDA];

  int t = threadIdx.x, lane = t & 31, w = t >> 5;
  int mW = (w >> 1) * 32, nW = (w & 1) * 32;

  auto stage = [&](int buf, int k0) {
#pragma unroll
    for (int p = 0; p < 2; ++p) {
      int idx = t + p * 256;
      int row = idx >> 2, c = idx & 3;
      cp16_async(&As[buf][row * LDA + c * 8],
                 shh + (size_t)(m0 + row) * SHH + k0 + c * 8);
    }
    {
      int row = t >> 2, c = t & 3;
      cp16_async(&Bs[buf][row * LDA + c * 8],
                 shdb + (size_t)(ntile + row) * SHH + k0 + c * 8);
    }
  };

  f32x8 zero = {0.f,0.f,0.f,0.f,0.f,0.f,0.f,0.f};
  f32x8 acc[2][2];
#pragma unroll
  for (int i = 0; i < 2; ++i)
#pragma unroll
    for (int j = 0; j < 2; ++j) acc[i][j] = zero;

  stage(0, 0);
  const int NIT = SHH / KT;
  for (int it = 0; it < NIT; ++it) {
    wait_async();
    __syncthreads();
    if (it + 1 < NIT) stage((it + 1) & 1, (it + 1) * KT);
    int b = it & 1;
    bf16x16 a0 = ldsFrag(As[b], mW +  0, lane);
    bf16x16 a1 = ldsFrag(As[b], mW + 16, lane);
    bf16x16 b0 = ldsFrag(Bs[b], nW +  0, lane);
    bf16x16 b1 = ldsFrag(Bs[b], nW + 16, lane);
    acc[0][0] = wmma_bf16(a0, b0, acc[0][0]);
    acc[0][1] = wmma_bf16(a0, b1, acc[0][1]);
    acc[1][0] = wmma_bf16(a1, b0, acc[1][0]);
    acc[1][1] = wmma_bf16(a1, b1, acc[1][1]);
  }

  int rbase = 8 * (lane >> 4);
  int nlo   = lane & 15;
#pragma unroll
  for (int i = 0; i < 2; ++i)
#pragma unroll
    for (int j = 0; j < 2; ++j) {
      int col = ntile + nW + j * 16 + nlo;
#pragma unroll
      for (int v = 0; v < 8; ++v) {
        int m = m0 + mW + i * 16 + rbase + v;
        out[(size_t)m * DIMX + col] = acc[i][j][v];
      }
    }
}

// =====================================================================
// Kernel 6: expert down GEMM + weighted atomic scatter into out
// grid: (DIMX/NT, CAPX/MT, NEXP), block 256
// =====================================================================
__global__ __launch_bounds__(256)
void k_down(const unsigned short* __restrict__ hidden,  // bf16 [NSLOT][EDIM]
            const unsigned short* __restrict__ dwt,     // bf16 [E][DIMX][EDIM]
            const int*   __restrict__ cnt,
            const int*   __restrict__ slotlist,
            const float* __restrict__ wslot,
            const unsigned short* __restrict__ zbuf,
            float* __restrict__ out) {
  int e     = blockIdx.z;
  int count = cnt[e];
  int m0    = blockIdx.y * MT;
  if (m0 >= count) return;
  int ntile = blockIdx.x * NT;

  __shared__ unsigned short As[2][MT * LDA];
  __shared__ unsigned short Bs[2][NT * LDA];
  __shared__ int slotS[MT];

  int t = threadIdx.x, lane = t & 31, w = t >> 5;
  int mW = (w >> 1) * 32, nW = (w & 1) * 32;

  if (t < MT) {
    int gm = m0 + t;
    slotS[t] = (gm < count) ? slotlist[e * CAPX + gm] : -1;
  }
  __syncthreads();

  const unsigned short* dwe = dwt + (size_t)e * DIMX * EDIM;

  auto stage = [&](int buf, int k0) {
#pragma unroll
    for (int p = 0; p < 2; ++p) {
      int idx = t + p * 256;
      int row = idx >> 2, c = idx & 3;
      int slot = slotS[row];
      const unsigned short* src = (slot >= 0)
          ? hidden + (size_t)slot * EDIM + k0 + c * 8
          : zbuf + c * 8;
      cp16_async(&As[buf][row * LDA + c * 8], src);
    }
    {
      int row = t >> 2, c = t & 3;
      cp16_async(&Bs[buf][row * LDA + c * 8],
                 dwe + (size_t)(ntile + row) * EDIM + k0 + c * 8);
    }
  };

  f32x8 zero = {0.f,0.f,0.f,0.f,0.f,0.f,0.f,0.f};
  f32x8 acc[2][2];
#pragma unroll
  for (int i = 0; i < 2; ++i)
#pragma unroll
    for (int j = 0; j < 2; ++j) acc[i][j] = zero;

  stage(0, 0);
  const int NIT = EDIM / KT;
  for (int it = 0; it < NIT; ++it) {
    wait_async();
    __syncthreads();
    if (it + 1 < NIT) stage((it + 1) & 1, (it + 1) * KT);
    int b = it & 1;
    bf16x16 a0 = ldsFrag(As[b], mW +  0, lane);
    bf16x16 a1 = ldsFrag(As[b], mW + 16, lane);
    bf16x16 b0 = ldsFrag(Bs[b], nW +  0, lane);
    bf16x16 b1 = ldsFrag(Bs[b], nW + 16, lane);
    acc[0][0] = wmma_bf16(a0, b0, acc[0][0]);
    acc[0][1] = wmma_bf16(a0, b1, acc[0][1]);
    acc[1][0] = wmma_bf16(a1, b0, acc[1][0]);
    acc[1][1] = wmma_bf16(a1, b1, acc[1][1]);
  }

  int rbase = 8 * (lane >> 4);
  int nlo   = lane & 15;
#pragma unroll
  for (int i = 0; i < 2; ++i)
#pragma unroll
    for (int j = 0; j < 2; ++j) {
      int col = ntile + nW + j * 16 + nlo;
#pragma unroll
      for (int v = 0; v < 8; ++v) {
        int m = mW + i * 16 + rbase + v;
        int slot = slotS[m];
        if (slot < 0) continue;
        int   token = slot >> 1;
        float val   = acc[i][j][v] * wslot[slot];
        __hip_atomic_fetch_add(&out[(size_t)token * DIMX + col], val,
                               __ATOMIC_RELAXED, __HIP_MEMORY_SCOPE_AGENT);
      }
    }
}

// =====================================================================
// host launcher
// =====================================================================
extern "C" void kernel_launch(void* const* d_in, const int* in_sizes, int n_in,
                              void* d_out, int out_size, void* d_ws, size_t ws_size,
                              hipStream_t stream) {
  (void)in_sizes; (void)n_in; (void)out_size; (void)ws_size;

  const float* x    = (const float*)d_in[0];
  const float* rw   = (const float*)d_in[1];
  const float* rb   = (const float*)d_in[2];
  const float* upw  = (const float*)d_in[3];
  const float* gw   = (const float*)d_in[4];
  const float* dwn  = (const float*)d_in[5];
  const float* shg  = (const float*)d_in[6];
  const float* shu  = (const float*)d_in[7];
  const float* shd  = (const float*)d_in[8];
  float* out = (float*)d_out;

  // ---- workspace layout ----
  char* ws = (char*)d_ws;
  int*      cnt      = (int*)ws;                  ws += 256;
  unsigned* zbuf     = (unsigned*)ws;             ws += 256;          // 128B used
  int*      slotlist = (int*)ws;                  ws += (size_t)NEXP * CAPX * 4;
  float*    wslot    = (float*)ws;                ws += (size_t)NSLOT * 4;
  unsigned short* xb   = (unsigned short*)ws;     ws += (size_t)NTOK * DIMX * 2;
  unsigned short* gwt  = (unsigned short*)ws;     ws += (size_t)NEXP * DIMX * EDIM * 2;
  unsigned short* uwt  = (unsigned short*)ws;     ws += (size_t)NEXP * DIMX * EDIM * 2;
  unsigned short* dwt  = (unsigned short*)ws;     ws += (size_t)NEXP * EDIM * DIMX * 2;
  unsigned short* shgb = (unsigned short*)ws;     ws += (size_t)SHH * DIMX * 2;
  unsigned short* shub = (unsigned short*)ws;     ws += (size_t)SHH * DIMX * 2;
  unsigned short* shdb = (unsigned short*)ws;     ws += (size_t)DIMX * SHH * 2;
  unsigned short* hidden = (unsigned short*)ws;   ws += (size_t)NSLOT * EDIM * 2;
  unsigned short* shh    = (unsigned short*)ws;   ws += (size_t)NTOK * SHH * 2;

  k_init  <<<1, 64, 0, stream>>>(cnt, zbuf);
  k_router<<<NTOK / 256, 256, 0, stream>>>(x, rw, rb, cnt, slotlist, wslot);

  // conversions
  k_cvt   <<<(NTOK * DIMX) / 2048, 256, 0, stream>>>(x,   xb,   (long)NTOK * DIMX);
  k_cvt   <<<(SHH * DIMX)  / 2048, 256, 0, stream>>>(shg, shgb, (long)SHH * DIMX);
  k_cvt   <<<(SHH * DIMX)  / 2048, 256, 0, stream>>>(shu, shub, (long)SHH * DIMX);
  k_cvt   <<<(DIMX * SHH)  / 2048, 256, 0, stream>>>(shd, shdb, (long)DIMX * SHH);
  k_cvt_t <<<dim3(EDIM / 64, DIMX / 64, NEXP), 256, 0, stream>>>(gw,  gwt, DIMX, EDIM);
  k_cvt_t <<<dim3(EDIM / 64, DIMX / 64, NEXP), 256, 0, stream>>>(upw, uwt, DIMX, EDIM);
  k_cvt_t <<<dim3(DIMX / 64, EDIM / 64, NEXP), 256, 0, stream>>>(dwn, dwt, EDIM, DIMX);

  // GEMMs
  k_upgate  <<<dim3(EDIM / NT, CAPX / MT, NEXP), 256, 0, stream>>>(
      xb, gwt, uwt, cnt, slotlist, (const unsigned short*)zbuf, hidden);
  k_shhidden<<<dim3(SHH / NT, NTOK / MT), 256, 0, stream>>>(xb, shgb, shub, shh);
  k_shdown  <<<dim3(DIMX / NT, NTOK / MT), 256, 0, stream>>>(shh, shdb, out);
  k_down    <<<dim3(DIMX / NT, CAPX / MT, NEXP), 256, 0, stream>>>(
      hidden, dwt, cnt, slotlist, wslot, (const unsigned short*)zbuf, out);
}
